// HausdorffDistanceLoss_48464410968519
// MI455X (gfx1250) — compile-verified, matrix operand order checked by reading, not run on hardware
//
#include <hip/hip_runtime.h>

// Problem constants (fixed by the reference's setup_inputs)
#define B_ 4
#define C_ 4
#define H_ 96
#define W_ 96
#define N_ (H_ * W_)
#define BIGF 1e10f

typedef __attribute__((ext_vector_type(16))) _Float16 v16h;
typedef __attribute__((ext_vector_type(8)))  float    v8f;
typedef unsigned uint32x4 __attribute__((ext_vector_type(4)));
typedef int      int32x8  __attribute__((ext_vector_type(8)));
typedef int      int32x4  __attribute__((ext_vector_type(4)));

union Frag { v16h h; unsigned u[8]; };

#if defined(__AMDGCN__) && __has_builtin(__builtin_amdgcn_tensor_load_to_lds) && \
    __has_builtin(__builtin_amdgcn_s_wait_tensorcnt)
#define HAVE_TDM 1
#if __has_include(<hip/amd_detail/amd_gfx1250_TDM.h>)
#define TDM_6ARG 1   // amdgpu-toolchain / therock-10.0 headers -> 6-arg builtin
#endif
#endif

// Monotone float<->uint mapping so atomicMax on unsigned == max on float
// (handles negatives: -1e10 sentinel for empty sets).
__device__ __forceinline__ unsigned mapOrder(float f) {
  unsigned u = __float_as_uint(f);
  return (u & 0x80000000u) ? ~u : (u | 0x80000000u);
}
__device__ __forceinline__ float unmapOrder(unsigned m) {
  return (m & 0x80000000u) ? __uint_as_float(m & 0x7FFFFFFFu)
                           : __uint_as_float(~m);
}

// ---------------- init: reset per-call state in workspace ----------------
__global__ void k_init(int* counts, unsigned* result) {
  int t = threadIdx.x;
  if (t < 8) {
    counts[t] = 0;
    result[t] = mapOrder(-BIGF);  // reference's -BIG when a set is empty
  }
}

// ---------------- argmax over channels (jnp.argmax: first max wins) ------
__global__ void k_argmax(const float* __restrict__ out, int* __restrict__ pred) {
  int idx = blockIdx.x * blockDim.x + threadIdx.x;  // over B*N
  if (idx >= B_ * N_) return;
  int b = idx / N_, p = idx - b * N_;
  const float* base = out + (size_t)b * C_ * N_ + p;
  float best = base[0];
  int bi = 0;
#pragma unroll
  for (int c = 1; c < C_; ++c) {
    float v = base[(size_t)c * N_];
    if (v > best) { best = v; bi = c; }
  }
  pred[idx] = bi;
}

// ---------------- boundary: any 4-neighbor with a different label --------
__global__ void k_boundary(const int* __restrict__ labels,
                           unsigned char* __restrict__ mask) {
  int idx = blockIdx.x * blockDim.x + threadIdx.x;  // over B*N
  if (idx >= B_ * N_) return;
  int p = idx % N_;
  int i = p / W_, j = p - i * W_;
  int v = labels[idx];
  bool bd = false;
  if (i > 0)      bd |= (labels[idx - W_] != v);
  if (i < H_ - 1) bd |= (labels[idx + W_] != v);
  if (j > 0)      bd |= (labels[idx - 1] != v);
  if (j < W_ - 1) bd |= (labels[idx + 1] != v);
  mask[idx] = bd ? 1 : 0;
}

// ---------------- compact boundary pixels into interleaved point lists ---
// pts[list][q] = { packed (f16 i | f16 j << 16), bits(i*i + j*j) }
__global__ void k_compact(const unsigned char* __restrict__ mask, int which,
                          int* __restrict__ counts, uint2* __restrict__ pts) {
  int idx = blockIdx.x * blockDim.x + threadIdx.x;  // over B*N
  if (idx >= B_ * N_) return;
  if (!mask[idx]) return;
  int b = idx / N_, p = idx - b * N_;
  int i = p / W_, j = p - i * W_;
  int list = b * 2 + which;
  int pos = atomicAdd(&counts[list], 1);
  _Float16 hi = (_Float16)(float)i;   // 0..95: exact in f16
  _Float16 hj = (_Float16)(float)j;
  unsigned short ui = __builtin_bit_cast(unsigned short, hi);
  unsigned short uj = __builtin_bit_cast(unsigned short, hj);
  uint2 e;
  e.x = (unsigned)ui | ((unsigned)uj << 16);          // K=0 low, K=1 high
  e.y = __float_as_uint((float)(i * i + j * j));
  pts[(size_t)list * N_ + pos] = e;
}

// ---------------- directed Hausdorff via WMMA cross dot products ---------
// blockIdx.x = b*2 + dir (dir0: A=pred,B=tgt ; dir1: A=tgt,B=pred)
// B list is staged into LDS once per block via the Tensor Data Mover, padded
// to a tile multiple with {0, +BIG} sentinels so the inner loop is branch-
// and clamp-free: one ds_load_b64 + one v_wmma per 16x16 tile pair.
__global__ void k_hausdorff(const int* __restrict__ counts,
                            const uint2* __restrict__ pts,
                            unsigned* __restrict__ result) {
  extern __shared__ unsigned ldsB[];   // [tilesB*16*2] staged (pk, sq) dwords

  const int pair = blockIdx.x;
  const int b = pair >> 1, dir = pair & 1;
  const int listA = b * 2 + dir;
  const int listB = b * 2 + (dir ^ 1);
  const int cA = counts[listA];
  const int cB = counts[listB];
  if (cA == 0 || cB == 0) return;  // block-uniform; result keeps -BIG sentinel

  const uint2* ptsA = pts + (size_t)listA * N_;
  const uint2* ptsB = pts + (size_t)listB * N_;

  const int lane = threadIdx.x & 31;   // gfx1250 is wave32
  const int wave = threadIdx.x >> 5;
  const int hi   = lane >> 4;          // D rows: M = r + 8*hi
  const int nl   = lane & 15;          // D col:  N = lane & 15

  const int tilesA = (cA + 15) >> 4;
  const int tilesB = (cB + 15) >> 4;
  const int padPts = tilesB * 16;

  // ---- stage B list into LDS ----
#if defined(HAVE_TDM)
  if (wave == 0) {
    unsigned nWords = (unsigned)(cB * 2);                    // dwords to move
    unsigned long long ga = (unsigned long long)(uintptr_t)ptsB;
    unsigned ldsOff = (unsigned)(uintptr_t)(void*)ldsB;      // flat low32 = LDS byte offset
    // D# group 0: count=1(valid), lds_addr, global_addr[56:0], type=2
    uint32x4 g0 = { 1u,
                    ldsOff,
                    (unsigned)(ga & 0xFFFFFFFFu),
                    (unsigned)((ga >> 32) & 0x01FFFFFFu) | (2u << 30) };
    // D# group 1: data_size=4B(code 2); tensor_dim0 = nWords; tensor_dim1 = 1;
    //             tile_dim0 = nWords (1 row); stride0 = nWords.
    int32x8 g1 = { (int)(2u << 16),
                   (int)((nWords & 0xFFFFu) << 16),                   // dim0[15:0]
                   (int)(((nWords >> 16) & 0xFFFFu) | (1u << 16)),    // dim0[31:16] | dim1=1
                   (int)((nWords & 0xFFFFu) << 16),                   // tile_dim0
                   0,
                   (int)nWords,                                       // dim0_stride lo
                   0, 0 };
    int32x4 gz = { 0, 0, 0, 0 };
#if defined(TDM_6ARG)
    int32x8 gz8 = { 0, 0, 0, 0, 0, 0, 0, 0 };
    __builtin_amdgcn_tensor_load_to_lds(g0, g1, gz, gz, gz8, 0);
#else
    __builtin_amdgcn_tensor_load_to_lds(g0, g1, gz, gz, 0);
#endif
  }
  // pad sentinel points (disjoint from TDM's write range)
  for (int q = cB + threadIdx.x; q < padPts; q += blockDim.x) {
    ldsB[2 * q]     = 0u;                       // coords (0,0): no dot contribution
    ldsB[2 * q + 1] = __float_as_uint(BIGF);    // |b|^2 = +BIG: loses every min
  }
  if (wave == 0) __builtin_amdgcn_s_wait_tensorcnt(0);
#else
  // fallback: cooperative staging
  const unsigned* src = (const unsigned*)ptsB;
  for (int w = threadIdx.x; w < cB * 2; w += blockDim.x) ldsB[w] = src[w];
  for (int q = cB + threadIdx.x; q < padPts; q += blockDim.x) {
    ldsB[2 * q]     = 0u;
    ldsB[2 * q + 1] = __float_as_uint(BIGF);
  }
#endif
  __syncthreads();

  const int wavesPerBlk = blockDim.x >> 5;
  const int waveG = blockIdx.y * wavesPerBlk + wave;
  const int waveS = gridDim.y * wavesPerBlk;

  for (int ta = waveG; ta < tilesA; ta += waveS) {
    // A fragment: 16-bit A layout -> lanes 0-15 carry M=lane, VGPR0 = K0,K1.
    int aIdx = ta * 16 + nl;
    uint2 pa = ptsA[min(aIdx, cA - 1)];
    unsigned apk = (lane < 16 && aIdx < cA) ? pa.x : 0u;
    Frag a;
#pragma unroll
    for (int k = 0; k < 8; ++k) a.u[k] = 0u;
    a.u[0] = apk;

    // |a|^2 for this lane's 8 rows; pad rows sink to -BIG (lose the max)
    float sa[8];
#pragma unroll
    for (int r = 0; r < 8; ++r) {
      int m = ta * 16 + 8 * hi + r;
      uint2 pm = ptsA[min(m, cA - 1)];
      sa[r] = (m < cA) ? __uint_as_float(pm.y) : -BIGF;
    }

    float rmin[8];
#pragma unroll
    for (int r = 0; r < 8; ++r) rmin[r] = BIGF;

    for (int tb = 0; tb < tilesB; ++tb) {
      unsigned base = (unsigned)(tb * 16 + nl) * 2u;
      unsigned bpk = ldsB[base];                         // one ds_load_b64 pair
      float sb = __uint_as_float(ldsB[base + 1]);
      bpk = (lane < 16) ? bpk : 0u;

      Frag bf;
#pragma unroll
      for (int k = 0; k < 8; ++k) bf.u[k] = 0u;
      bf.u[0] = bpk;

      v8f c = {};
      v8f d = __builtin_amdgcn_wmma_f32_16x16x32_f16(
          /*neg_a=*/false, a.h, /*neg_b=*/false, bf.h,
          /*c_mod=*/(short)0, c, /*reuse_a=*/false, /*reuse_b=*/false);

#pragma unroll
      for (int r = 0; r < 8; ++r) {
        float d2 = sa[r] + sb - 2.0f * d[r];   // (ia-ib)^2 + (ja-jb)^2, exact
        rmin[r] = fminf(rmin[r], d2);
      }
    }

    // min over the 16 columns spread across lanes of each half
#pragma unroll
    for (int m = 1; m <= 8; m <<= 1) {
#pragma unroll
      for (int r = 0; r < 8; ++r)
        rmin[r] = fminf(rmin[r], __shfl_xor(rmin[r], m, 32));
    }
    // max over this half's 8 rows, then combine halves (rows 0-7 vs 8-15)
    float mx = rmin[0];
#pragma unroll
    for (int r = 1; r < 8; ++r) mx = fmaxf(mx, rmin[r]);
    mx = fmaxf(mx, __shfl_xor(mx, 16, 32));
    if (lane == 0) atomicMax(&result[pair], mapOrder(mx));
  }
}

// ---------------- finalize: sqrt(max(fwd,bwd,0)) averaged over batch -----
__global__ void k_final(const unsigned* __restrict__ result, float* out) {
  float acc = 0.0f;
#pragma unroll
  for (int b = 0; b < B_; ++b) {
    float fwd = unmapOrder(result[b * 2 + 0]);
    float bwd = unmapOrder(result[b * 2 + 1]);
    float h = fmaxf(fmaxf(fwd, bwd), 0.0f);
    acc += sqrtf(h);
  }
  out[0] = acc / (float)B_;
}

extern "C" void kernel_launch(void* const* d_in, const int* in_sizes, int n_in,
                              void* d_out, int out_size, void* d_ws, size_t ws_size,
                              hipStream_t stream) {
  const float* output = (const float*)d_in[0];  // [B,C,H,W] f32
  const int*   target = (const int*)d_in[1];    // [B,H,W] i32

  // Workspace layout (bytes)
  char* ws = (char*)d_ws;
  int*           counts = (int*)(ws + 0);             // 8 ints
  unsigned*      result = (unsigned*)(ws + 32);       // 8 uints
  int*           pred   = (int*)(ws + 64);            // B*N ints
  unsigned char* maskP  = (unsigned char*)(ws + 64 + (size_t)B_ * N_ * 4);
  unsigned char* maskT  = maskP + (size_t)B_ * N_;
  uint2*         pts    = (uint2*)((char*)maskT + (size_t)B_ * N_);  // 8 lists * N_ uint2

  const int threads = 256;
  const int nBN = (B_ * N_ + threads - 1) / threads;
  const size_t ldsBytes = (size_t)N_ * 8;  // worst case: 9216 pts * 8B = 72KB (<320KB/WGP)

  k_init<<<1, 8, 0, stream>>>(counts, result);
  k_argmax<<<nBN, threads, 0, stream>>>(output, pred);
  k_boundary<<<nBN, threads, 0, stream>>>(pred, maskP);
  k_boundary<<<nBN, threads, 0, stream>>>(target, maskT);
  k_compact<<<nBN, threads, 0, stream>>>(maskP, 0, counts, pts);
  k_compact<<<nBN, threads, 0, stream>>>(maskT, 1, counts, pts);
  k_hausdorff<<<dim3(B_ * 2, 12), threads, ldsBytes, stream>>>(counts, pts, result);
  k_final<<<1, 1, 0, stream>>>(result, (float*)d_out);
}
